// VectorQuantizer_19851338842614
// MI455X (gfx1250) — matrix operand order repaired and checked by reading, not compiled, not run on hardware
//
#include <hip/hip_runtime.h>
#include <hip/hip_bf16.h>

// ---------------- problem constants ----------------
#define NUM_CODES 1024
#define CODE_DIM  256
#define T_TOTAL   32768            // 32*32*32 spatial vectors
#define M_TILE    64               // rows of z_flat per workgroup
#define NUM_WG    (T_TOTAL / M_TILE)   // 512
#define ZQ_ELEMS  8388608          // 32*256*32*32
#define IDX_OFF   ZQ_ELEMS
#define LOSS_OFF  (ZQ_ELEMS + T_TOTAL)
#define PPL_OFF   (ZQ_ELEMS + T_TOTAL + 1)

// workspace layout (bytes)
#define WS_BFB    0                // bf16 codebook fragments: 1024*256*2 = 524288
#define WS_ESQ    524288           // 1024 f32
#define WS_CNT    528384           // 1024 i32
#define WS_PART   532480           // 512 f32 partial loss sums

typedef __attribute__((ext_vector_type(16))) __bf16 v16bf;
typedef __attribute__((ext_vector_type(8)))  __bf16 v8bf;
typedef __attribute__((ext_vector_type(8)))  float  v8f;

struct A2 { v8bf lo, hi; };

__device__ __forceinline__ unsigned short f2bf(float f) {
    union { float f; unsigned u; } x; x.f = f;
    unsigned u = x.u;
    u += 0x7FFFu + ((u >> 16) & 1u);           // round-to-nearest-even
    return (unsigned short)(u >> 16);
}

// ---------------------------------------------------------------------------
// Kernel 1: codebook -> bf16 B-fragment layout, e_sq, zero histogram.
// B fragment for (n-tile nt, k-step ks): lane l<16 holds code n0+l, channels
// k0..k0+15 contiguous; lane l>=16 holds code n0+l-16, channels k0+16..k0+31.
// Each lane's 16 bf16 (32B) contiguous at halfIdx = ((nt*8+ks)*32+lane)*16+j.
// ---------------------------------------------------------------------------
__global__ __launch_bounds__(256) void vq_prep(const float* __restrict__ cb,
                                               unsigned short* __restrict__ bfB,
                                               float* __restrict__ esq,
                                               int* __restrict__ counts) {
    __shared__ float red[256];
    const int n = blockIdx.x;      // code id
    const int c = threadIdx.x;     // channel
    float v = cb[n * CODE_DIM + c];

    const int nt = n >> 4, l0 = n & 15;
    const int ks = c >> 5, kk = c & 31;
    const int lane = (kk < 16) ? l0 : (l0 + 16);
    const int j = kk & 15;
    bfB[((((nt << 3) + ks) * 32 + lane) << 4) + j] = f2bf(v);

    red[c] = v * v;
    __syncthreads();
#pragma unroll
    for (int s = 128; s > 0; s >>= 1) {
        if (c < s) red[c] += red[c + s];
        __syncthreads();
    }
    if (c == 0) { esq[n] = red[0]; counts[n] = 0; }
}

// ---------------------------------------------------------------------------
// Kernel 2: main. 512 WGs x 64 threads (2 waves). Each WG owns 64 rows of
// z_flat = 2 (b,h) pairs x 32 w. Each wave owns 32 rows = 2 A-tiles and scans
// all 1024 codes; B fragments are register double-buffered across n-tiles so
// the v_wmma stream never waits on loadcnt for the current tile.
// ---------------------------------------------------------------------------
__global__ __launch_bounds__(64) void vq_main(const float* __restrict__ z,
                                              const float* __restrict__ cb,
                                              const unsigned short* __restrict__ bfB,
                                              const float* __restrict__ esq,
                                              int* __restrict__ counts,
                                              float* __restrict__ partial,
                                              float* __restrict__ out) {
    // padded pitch: 264 halves = 528 B = 132 dwords -> lane bank stride 4 (conflict-free b128)
    __shared__ unsigned short sZ[M_TILE * 264];
    __shared__ int   sIdx[M_TILE];
    __shared__ float sRed[64];

    const int wg  = blockIdx.x;
    const int tid = threadIdx.x;
    const int x   = tid & 31;        // lane / w coordinate
    const int y   = tid >> 5;        // wave id 0..1

    // ---- stage 1: z tile [64 rows x 256 ch] fp32 -> bf16 LDS (coalesced on w) ----
#pragma unroll
    for (int p = 0; p < 2; ++p) {
        const int P = wg * 2 + p;            // global (b,h) pair
        const int b = P >> 5, h = P & 31;
        const int rowbase = (p * 32 + x) * 264;
        const long zbase = (long)b * 262144 + (long)h * 32 + x;
#pragma unroll 8
        for (int c0 = 0; c0 < CODE_DIM; c0 += 2) {
            const int c = c0 + y;
            sZ[rowbase + c] = f2bf(z[zbase + (long)c * 1024]);
        }
    }
    __syncthreads();

    // ---- per-wave WMMA argmax over codes ----
    const int lane = x;
    const int halfsel = lane >> 4;               // 0: lanes 0-15, 1: lanes 16-31
    const int m0 = y * 32;                       // wave's first local row

    // A fragments for 2 tiles (rows m0..m0+15, m0+16..m0+31).
    // lane<16 : e0..15 = K k0..k0+15
    // lane>=16: e0..15 = K k0+16..k0+31 (matches B fragment layout)
    v16bf aF[2][8];
#pragma unroll
    for (int t = 0; t < 2; ++t) {
        const unsigned short* rp = &sZ[(m0 + t * 16 + (lane & 15)) * 264];
#pragma unroll
        for (int ks = 0; ks < 8; ++ks) {
            A2 a;
            a.lo = *(const v8bf*)(rp + ks * 32 + (halfsel ? 16 : 0));
            a.hi = *(const v8bf*)(rp + ks * 32 + 8 + (halfsel ? 16 : 0));
            aF[t][ks] = __builtin_bit_cast(v16bf, a);
        }
    }

    float bestV[2][8];
    int   bestI[2][8];
#pragma unroll
    for (int t = 0; t < 2; ++t)
#pragma unroll
        for (int j = 0; j < 8; ++j) { bestV[t][j] = -3.4e38f; bestI[t][j] = 0; }

    auto loadB = [&](v16bf* b, int nt) {
#pragma unroll
        for (int ks = 0; ks < 8; ++ks)
            b[ks] = *(const v16bf*)(bfB + ((((nt << 3) + ks) * 32 + lane) << 4));
    };
    auto compute = [&](int nt, const v16bf* b) {
        v8f acc0 = {}, acc1 = {};
#pragma unroll
        for (int ks = 0; ks < 8; ++ks) {
            // two independent accumulation chains sharing one B fragment
            acc0 = __builtin_amdgcn_wmma_f32_16x16x32_bf16(
                false, aF[0][ks], false, b[ks], (short)0, acc0, false, false);
            acc1 = __builtin_amdgcn_wmma_f32_16x16x32_bf16(
                false, aF[1][ks], false, b[ks], (short)0, acc1, false, false);
        }
        const int   n  = (nt << 4) + (lane & 15);     // column (code id)
        const float he = 0.5f * esq[n];
#pragma unroll
        for (int j = 0; j < 8; ++j) {
            const float v0 = acc0[j] - he;            // = z.e - ||e||^2/2
            if (v0 > bestV[0][j]) { bestV[0][j] = v0; bestI[0][j] = n; }
            const float v1 = acc1[j] - he;
            if (v1 > bestV[1][j]) { bestV[1][j] = v1; bestI[1][j] = n; }
        }
    };

    v16bf b0[8], b1[8];
    loadB(b0, 0);
    for (int nt = 0; nt < 64; nt += 2) {
        loadB(b1, nt + 1);                       // prefetch odd tile into regs
        __builtin_prefetch(bfB + (((nt + 4) & 63) << 12), 0, 0);  // L2->WGP$
        compute(nt, b0);                         // consume even tile
        loadB(b0, (nt + 2) & 63);                // prefetch next even tile
        compute(nt + 1, b1);                     // consume odd tile
    }

    // reduce across the 16 lanes holding the same rows (xor 1,2,4,8 stays in half)
#pragma unroll
    for (int mask = 1; mask <= 8; mask <<= 1) {
#pragma unroll
        for (int t = 0; t < 2; ++t)
#pragma unroll
            for (int j = 0; j < 8; ++j) {
                const float ov = __shfl_xor(bestV[t][j], mask, 32);
                const int   oi = __shfl_xor(bestI[t][j], mask, 32);
                if (ov > bestV[t][j] || (ov == bestV[t][j] && oi < bestI[t][j])) {
                    bestV[t][j] = ov; bestI[t][j] = oi;
                }
            }
    }
    // C/D layout: VGPR j = row base+j (lanes 0-15) or base+8+j (lanes 16-31)
    if ((lane & 15) == 0) {
#pragma unroll
        for (int t = 0; t < 2; ++t) {
            const int rbase = m0 + t * 16 + (halfsel ? 8 : 0);
#pragma unroll
            for (int j = 0; j < 8; ++j) sIdx[rbase + j] = bestI[t][j];
        }
    }
    __syncthreads();

    // ---- histogram + index output ----
    {
        const int idx = sIdx[tid];
        atomicAdd(&counts[idx], 1);
        out[IDX_OFF + wg * M_TILE + tid] = (float)idx;
    }

    // ---- z_q gather/store (coalesced on w) + squared-error partial ----
    float lsum = 0.f;
#pragma unroll
    for (int p = 0; p < 2; ++p) {
        const int P = wg * 2 + p;
        const int b = P >> 5, h = P & 31;
        const int idx = sIdx[p * 32 + x];
        const float* crow = cb + (long)idx * CODE_DIM;
        const long zbase = (long)b * 262144 + (long)h * 32 + x;
#pragma unroll 8
        for (int c0 = 0; c0 < CODE_DIM; c0 += 2) {
            const int c = c0 + y;
            const long zi = zbase + (long)c * 1024;
            const float q  = crow[c];
            const float zv = z[zi];
            out[zi] = q;                       // z_q_st == z_q numerically
            const float d = q - zv;
            lsum += d * d;
        }
    }
    sRed[tid] = lsum;
    __syncthreads();
#pragma unroll
    for (int s = 32; s > 0; s >>= 1) {
        if (tid < s) sRed[tid] += sRed[tid + s];
        __syncthreads();
    }
    if (tid == 0) partial[wg] = sRed[0];
}

// ---------------------------------------------------------------------------
// Kernel 3: deterministic final reductions -> vq_loss, perplexity
// ---------------------------------------------------------------------------
__global__ __launch_bounds__(1024) void vq_final(const int* __restrict__ counts,
                                                 const float* __restrict__ partial,
                                                 float* __restrict__ out) {
    __shared__ float s[1024];
    const int t = threadIdx.x;
    const float p = (float)counts[t] * (1.0f / (float)T_TOTAL);
    s[t] = p * logf(p + 1e-10f);
    __syncthreads();
#pragma unroll
    for (int k = 512; k > 0; k >>= 1) {
        if (t < k) s[t] += s[t + k];
        __syncthreads();
    }
    const float H = s[0];
    __syncthreads();
    s[t] = (t < NUM_WG) ? partial[t] : 0.f;
    __syncthreads();
#pragma unroll
    for (int k = 512; k > 0; k >>= 1) {
        if (t < k) s[t] += s[t + k];
        __syncthreads();
    }
    if (t == 0) {
        out[PPL_OFF]  = expf(-H);
        // vq_loss = mean + BETA*mean = 1.25 * sum / numel
        out[LOSS_OFF] = 1.25f * s[0] / (float)ZQ_ELEMS;
    }
}

extern "C" void kernel_launch(void* const* d_in, const int* in_sizes, int n_in,
                              void* d_out, int out_size, void* d_ws, size_t ws_size,
                              hipStream_t stream) {
    const float* z  = (const float*)d_in[0];   // [32,256,32,32]
    const float* cb = (const float*)d_in[1];   // [1024,256]
    char* ws = (char*)d_ws;
    unsigned short* bfB   = (unsigned short*)(ws + WS_BFB);
    float*          esq   = (float*)(ws + WS_ESQ);
    int*            counts= (int*)(ws + WS_CNT);
    float*          part  = (float*)(ws + WS_PART);
    float* out = (float*)d_out;

    vq_prep <<<NUM_CODES, 256, 0, stream>>>(cb, bfB, esq, counts);
    vq_main <<<NUM_WG,     64, 0, stream>>>(z, cb, bfB, esq, counts, part, out);
    vq_final<<<1,        1024, 0, stream>>>(counts, part, out);
}
